// MattingCNN_22462678958637
// MI455X (gfx1250) — compile-verified, matrix operand order checked by reading, not run on hardware
//
#include <hip/hip_runtime.h>
#include <hip/hip_bf16.h>
#include <math.h>

typedef __attribute__((ext_vector_type(16))) _Float16 v16h;
typedef __attribute__((ext_vector_type(8)))  _Float16 v8h;
typedef __attribute__((ext_vector_type(8)))  float    v8f;

#define TB 256
#define CG_STEPS 50

// ---------------------------------------------------------------------------
// f32 -> f16 convert
__global__ void k_to_f16(const float* __restrict__ in, _Float16* __restrict__ out, int n) {
    int i = blockIdx.x * blockDim.x + threadIdx.x;
    if (i < n) out[i] = (_Float16)in[i];
}

// zero floats
__global__ void k_zero(float* __restrict__ p, int n) {
    int i = blockIdx.x * blockDim.x + threadIdx.x;
    if (i < n) p[i] = 0.f;
}

// ---------------------------------------------------------------------------
// Pack conv weights (OIHW f32) into WMMA B layout, f16.
// Layout: Bpk[((nt*KC + kc)*32 + lane)*16 + e]
// lane column n = nt*16 + (lane&15); K index = kc*32 + e + (e<8?0:8) + (lane>=16?8:0)
__global__ void k_pack_w(const float* __restrict__ w, _Float16* __restrict__ Bpk,
                         int Cin, int Cout, int KC, int nTiles) {
    int idx = blockIdx.x * blockDim.x + threadIdx.x;
    int total = nTiles * KC * 512;
    if (idx >= total) return;
    int e    = idx & 15;
    int lane = (idx >> 4) & 31;
    int t    = idx >> 9;
    int kc   = t % KC;
    int nt   = t / KC;
    int k = kc * 32 + e + (e < 8 ? 0 : 8) + ((lane >> 4) ? 8 : 0);
    int n = nt * 16 + (lane & 15);
    float v = 0.f;
    if (n < Cout && k < Cin * 9) {
        int cin = k / 9;
        int tap = k - cin * 9;
        v = w[((size_t)n * Cin + cin) * 9 + tap];
    }
    Bpk[idx] = (_Float16)v;
}

// ---------------------------------------------------------------------------
// Implicit-GEMM 3x3 SAME conv via WMMA f32_16x16x32_f16.
// Block = NT waves; each wave owns one 16-outchannel tile of the same
// 16-pixel row segment. The im2col slab (16 px x CIN*9 K, padded to KC*32)
// is staged once per block into LDS as lds[px][k] with pitch KROW+8 halves:
// - per-lane A chunk = two contiguous 16B groups -> ds_load_b128 x2
// - lane stride 2*(KROW+8) B keeps 16 lanes on distinct LDS banks
template<int CIN, int KC, int NT>
__global__ __launch_bounds__(NT * 32)
void k_conv_wmma(const _Float16* __restrict__ in16,
                 const _Float16* __restrict__ Bpk,
                 const float* __restrict__ bias,
                 float* __restrict__ out,
                 int H, int W, int Cout) {
    constexpr int KROW  = KC * 32;
    constexpr int PITCH = KROW + 8;         // halves; *2 bytes stays 16B aligned
    constexpr int KREAL = CIN * 9;
    __shared__ _Float16 lds[16 * PITCH];

    const int Np      = H * W;
    const int pixTile = blockIdx.x;
    const int tid     = threadIdx.x;
    const int pbase   = pixTile * 16;       // 16 pixels, all in one image row
    const int prow    = pbase / W;
    const int cbase   = pbase - prow * W;

    // ---- stage im2col tile into LDS ----
    for (int t = tid; t < 16 * KROW; t += NT * 32) {
        int px = t / KROW;
        int k  = t - px * KROW;
        _Float16 v = (_Float16)0.f;
        if (k < KREAL) {
            int cin = k / 9;
            int tap = k - cin * 9;
            int rr  = prow + tap / 3 - 1;
            int cc  = cbase + px + (tap - (tap / 3) * 3) - 1;
            if (rr >= 0 && rr < H && cc >= 0 && cc < W)
                v = in16[(size_t)cin * Np + rr * W + cc];
        }
        lds[px * PITCH + k] = v;
    }
    __syncthreads();

    const int wid  = tid >> 5;              // output-channel tile
    const int lane = tid & 31;
    const int l15  = lane & 15;
    const int hi   = lane >> 4;
    const int ab   = l15 * PITCH + hi * 8;  // A base (halves)

    v8f acc = {};
#pragma unroll
    for (int kc = 0; kc < KC; ++kc) {
        // A layout: e<8 -> K = kc*32 + hi*8 + e ; e>=8 -> K = kc*32+16 + hi*8 + (e-8)
        v8h alo = *(const v8h*)(&lds[ab + kc * 32]);
        v8h ahi = *(const v8h*)(&lds[ab + kc * 32 + 16]);
        v16h a;
#pragma unroll
        for (int e = 0; e < 8; ++e) { a[e] = alo[e]; a[e + 8] = ahi[e]; }
        v16h b = *(const v16h*)(Bpk + (((size_t)wid * KC + kc) * 32 + lane) * 16);
        acc = __builtin_amdgcn_wmma_f32_16x16x32_f16(false, a, false, b,
                                                     (short)0, acc, false, false);
    }
    // D layout: lane column n = l15 (out channel), rows m = r + 8*hi (pixel)
    int ch = wid * 16 + l15;
    if (ch < Cout) {
        float bb = bias[ch];
#pragma unroll
        for (int r = 0; r < 8; ++r)
            out[(size_t)ch * Np + pbase + r + hi * 8] = acc[r] + bb;
    }
}

// ---------------------------------------------------------------------------
// BatchNorm stats: one block per channel
__global__ __launch_bounds__(TB)
void k_bn_stats(const float* __restrict__ y, float* __restrict__ mean,
                float* __restrict__ inv, int Np) {
    __shared__ float ss[TB], sq[TB];
    int c = blockIdx.x;
    float s = 0.f, s2 = 0.f;
    for (int i = threadIdx.x; i < Np; i += TB) {
        float v = y[(size_t)c * Np + i];
        s += v; s2 += v * v;
    }
    ss[threadIdx.x] = s; sq[threadIdx.x] = s2;
    __syncthreads();
    for (int off = TB / 2; off > 0; off >>= 1) {
        if ((int)threadIdx.x < off) {
            ss[threadIdx.x] += ss[threadIdx.x + off];
            sq[threadIdx.x] += sq[threadIdx.x + off];
        }
        __syncthreads();
    }
    if (threadIdx.x == 0) {
        float m   = ss[0] / (float)Np;
        float var = sq[0] / (float)Np - m * m;
        mean[c] = m;
        inv[c]  = rsqrtf(var + 1e-5f);
    }
}

// BN apply + leaky relu (0.01) + convert to f16 for next conv. blockIdx.y = channel
__global__ void k_bn_act(const float* __restrict__ y, const float* __restrict__ mean,
                         const float* __restrict__ inv, const float* __restrict__ g,
                         const float* __restrict__ bta, _Float16* __restrict__ out16,
                         int Np) {
    int c = blockIdx.y;
    int i = blockIdx.x * blockDim.x + threadIdx.x;
    if (i >= Np) return;
    size_t idx = (size_t)c * Np + i;
    float v = (y[idx] - mean[c]) * inv[c] * g[c] + bta[c];
    v = v > 0.f ? v : 0.01f * v;
    out16[idx] = (_Float16)v;
}

// ---------------------------------------------------------------------------
// Sparse operator setup
__global__ void k_cm_setup(const int* __restrict__ row, const float* __restrict__ data,
                           const float* __restrict__ CMw, float* __restrict__ vals,
                           float* __restrict__ rs, int nnz) {
    int i = blockIdx.x * blockDim.x + threadIdx.x;
    if (i >= nnz) return;
    int r = row[i];
    float v = CMw[r] * data[i];
    vals[i] = v;
    atomicAdd(&rs[r], v);
}

// offs[j] = (j/3 - 1) + (j%3 - 1)*W ; row = ii + offs[0] = ii - 1 - W
__global__ void k_loc_setup(const int* __restrict__ inInd, const float* __restrict__ flows,
                            const float* __restrict__ LOCw, float* __restrict__ vals,
                            float* __restrict__ rs, int Kloc, int W) {
    int idx = blockIdx.x * blockDim.x + threadIdx.x;
    if (idx >= Kloc * 9) return;
    int k = idx / 9, j = idx - 9 * k;
    int ii = inInd[k];
    // vals[k,j] = LOC_flows[j,0,k] * LOC_w[ii]
    float v = flows[(size_t)j * 9 * Kloc + k] * LOCw[ii];
    int r  = ii - 1 - W;
    int cc = ii + (j / 3 - 1) + (j - (j / 3) * 3 - 1) * W;
    vals[idx] = v;
    atomicAdd(&rs[r],  0.5f * v);
    atomicAdd(&rs[cc], 0.5f * v);
}

__global__ void k_iu_setup(const int* __restrict__ inInd, const float* __restrict__ flows,
                           const int* __restrict__ neighInd, const float* __restrict__ IUw,
                           float* __restrict__ vals, float* __restrict__ rs, int Kiu) {
    int idx = blockIdx.x * blockDim.x + threadIdx.x;
    if (idx >= Kiu * 5) return;
    int k = idx / 5;
    int r = inInd[k];
    float v = flows[idx] * IUw[r];
    int c = neighInd[idx];
    vals[idx] = v;
    atomicAdd(&rs[r], 0.5f * v);
    atomicAdd(&rs[c], 0.5f * v);
}

// ---------------------------------------------------------------------------
// CG kernels. scal[0]=rs_old, scal[1]=pAp, scal[2]=rs_new
__global__ __launch_bounds__(TB)
void k_cg_init(const float* __restrict__ KUw, const float* __restrict__ kToUconf,
               const float* __restrict__ known, const float* __restrict__ kToU,
               const float* __restrict__ lmbda, float* __restrict__ diag,
               float* __restrict__ x, float* __restrict__ r, float* __restrict__ p,
               float* __restrict__ scal, int n) {
    __shared__ float red[TB];
    int i = blockIdx.x * blockDim.x + threadIdx.x;
    float contrib = 0.f;
    if (i < n) {
        float d  = KUw[i] * kToUconf[i] + lmbda[0] * known[i];
        float bb = d * kToU[i];
        diag[i] = d;
        x[i] = 0.f; r[i] = bb; p[i] = bb;
        contrib = bb * bb;
    }
    red[threadIdx.x] = contrib;
    __syncthreads();
    for (int off = TB / 2; off > 0; off >>= 1) {
        if ((int)threadIdx.x < off) red[threadIdx.x] += red[threadIdx.x + off];
        __syncthreads();
    }
    if (threadIdx.x == 0) atomicAdd(&scal[2], red[0]);
}

__global__ void k_iter_begin(float* __restrict__ S, float* __restrict__ scal, int n) {
    int i = blockIdx.x * blockDim.x + threadIdx.x;
    if (i < n) S[i] = 0.f;
    if (i == 0) { scal[0] = scal[2]; scal[1] = 0.f; scal[2] = 0.f; }
}

// S[row[i]] += vals[i] * v[col[i]]   (swap row/col args for the transpose)
__global__ void k_scatter_cm(const int* __restrict__ row, const int* __restrict__ col,
                             const float* __restrict__ vals, const float* __restrict__ v,
                             float* __restrict__ S, int nnz) {
    int i = blockIdx.x * blockDim.x + threadIdx.x;
    if (i >= nnz) return;
    atomicAdd(&S[row[i]], vals[i] * v[col[i]]);
}

// z = rs_cm*p - S ; S = 0
__global__ void k_zmk(const float* __restrict__ rs_cm, const float* __restrict__ p,
                      float* __restrict__ S, float* __restrict__ z, int n) {
    int i = blockIdx.x * blockDim.x + threadIdx.x;
    if (i >= n) return;
    z[i] = rs_cm[i] * p[i] - S[i];
    S[i] = 0.f;
}

__global__ void k_scatter_mat(const int* __restrict__ inInd, const float* __restrict__ vals,
                              const float* __restrict__ v, float* __restrict__ S,
                              int Kloc, int W) {
    int idx = blockIdx.x * blockDim.x + threadIdx.x;
    if (idx >= Kloc * 9) return;
    int k = idx / 9, j = idx - 9 * k;
    int ii = inInd[k];
    int r  = ii - 1 - W;
    int cc = ii + (j / 3 - 1) + (j - (j / 3) * 3 - 1) * W;
    float lv = 0.5f * vals[idx];
    atomicAdd(&S[r],  lv * v[cc]);
    atomicAdd(&S[cc], lv * v[r]);
}

__global__ void k_scatter_cs(const int* __restrict__ inInd, const int* __restrict__ neighInd,
                             const float* __restrict__ vals, const float* __restrict__ v,
                             float* __restrict__ S, int Kiu) {
    int idx = blockIdx.x * blockDim.x + threadIdx.x;
    if (idx >= Kiu * 5) return;
    int k = idx / 5;
    int r = inInd[k];
    int c = neighInd[idx];
    float lv = 0.5f * vals[idx];
    atomicAdd(&S[r], lv * v[c]);
    atomicAdd(&S[c], lv * v[r]);
}

// Ap = rs_cm*z + (rs_mat+rs_cs+diag)*p - S ; pAp -> scal[1]
__global__ __launch_bounds__(TB)
void k_combine(const float* __restrict__ rs_cm, const float* __restrict__ rs_mat,
               const float* __restrict__ rs_cs, const float* __restrict__ diag,
               const float* __restrict__ z, const float* __restrict__ p,
               const float* __restrict__ S, float* __restrict__ Ap,
               float* __restrict__ scal, int n) {
    __shared__ float red[TB];
    int i = blockIdx.x * blockDim.x + threadIdx.x;
    float contrib = 0.f;
    if (i < n) {
        float ap = rs_cm[i] * z[i] + (rs_mat[i] + rs_cs[i] + diag[i]) * p[i] - S[i];
        Ap[i] = ap;
        contrib = p[i] * ap;
    }
    red[threadIdx.x] = contrib;
    __syncthreads();
    for (int off = TB / 2; off > 0; off >>= 1) {
        if ((int)threadIdx.x < off) red[threadIdx.x] += red[threadIdx.x + off];
        __syncthreads();
    }
    if (threadIdx.x == 0) atomicAdd(&scal[1], red[0]);
}

// alpha = rs/(pAp+eps); x += alpha p; r -= alpha Ap; rr -> scal[2]
__global__ __launch_bounds__(TB)
void k_update1(const float* __restrict__ Ap, const float* __restrict__ p,
               float* __restrict__ x, float* __restrict__ r,
               float* __restrict__ scal, int n) {
    __shared__ float red[TB];
    float alpha = scal[0] / (scal[1] + 1e-12f);
    int i = blockIdx.x * blockDim.x + threadIdx.x;
    float contrib = 0.f;
    if (i < n) {
        x[i] += alpha * p[i];
        float rr = r[i] - alpha * Ap[i];
        r[i] = rr;
        contrib = rr * rr;
    }
    red[threadIdx.x] = contrib;
    __syncthreads();
    for (int off = TB / 2; off > 0; off >>= 1) {
        if ((int)threadIdx.x < off) red[threadIdx.x] += red[threadIdx.x + off];
        __syncthreads();
    }
    if (threadIdx.x == 0) atomicAdd(&scal[2], red[0]);
}

// beta = rs_new/(rs_old+eps); p = r + beta p
__global__ void k_update2(const float* __restrict__ r, float* __restrict__ p,
                          const float* __restrict__ scal, int n) {
    float beta = scal[2] / (scal[0] + 1e-12f);
    int i = blockIdx.x * blockDim.x + threadIdx.x;
    if (i >= n) return;
    p[i] = r[i] + beta * p[i];
}

// ---------------------------------------------------------------------------
extern "C" void kernel_launch(void* const* d_in, const int* in_sizes, int n_in,
                              void* d_out, int out_size, void* d_ws, size_t ws_size,
                              hipStream_t stream) {
    const float* image       = (const float*)d_in[0];
    const float* conv1_w     = (const float*)d_in[1];
    const float* conv1_b     = (const float*)d_in[2];
    const float* bn1_g       = (const float*)d_in[3];
    const float* bn1_b       = (const float*)d_in[4];
    const float* conv2_w     = (const float*)d_in[5];
    const float* conv2_b     = (const float*)d_in[6];
    const float* bn2_g       = (const float*)d_in[7];
    const float* bn2_b       = (const float*)d_in[8];
    const float* conv3_w     = (const float*)d_in[9];
    const float* conv3_b     = (const float*)d_in[10];
    const float* lmbda       = (const float*)d_in[11];
    const int*   Wcm_row     = (const int*)d_in[12];
    const int*   Wcm_col     = (const int*)d_in[13];
    const float* Wcm_data    = (const float*)d_in[14];
    const int*   LOC_inInd   = (const int*)d_in[15];
    const float* LOC_flows   = (const float*)d_in[16];
    const int*   IU_inInd    = (const int*)d_in[17];
    const float* IU_flows    = (const float*)d_in[18];
    const int*   IU_neighInd = (const int*)d_in[19];
    const float* kToUconf    = (const float*)d_in[20];
    const float* known       = (const float*)d_in[21];
    const float* kToU        = (const float*)d_in[22];

    const int Npix = in_sizes[0] / 3;            // h*w = 147456
    const int W    = 384;
    const int H    = Npix / W;
    const int NNZ  = in_sizes[12];
    const int Kloc = in_sizes[15];
    const int Kiu  = in_sizes[17];

    // bump allocator over workspace
    char*  base = (char*)d_ws;
    size_t off  = 0;
    auto alloc = [&](size_t bytes) -> void* {
        void* p = base + off;
        off = (off + bytes + 255) & ~(size_t)255;
        return p;
    };

    _Float16* act16 = (_Float16*)alloc((size_t)64 * Npix * sizeof(_Float16));
    float*    yb    = (float*)alloc((size_t)64 * Npix * sizeof(float));
    float*    w4    = (float*)alloc((size_t)4 * Npix * sizeof(float));
    _Float16* Bpk1  = (_Float16*)alloc((size_t)4 * 1  * 512 * sizeof(_Float16));
    _Float16* Bpk2  = (_Float16*)alloc((size_t)4 * 18 * 512 * sizeof(_Float16));
    _Float16* Bpk3  = (_Float16*)alloc((size_t)1 * 18 * 512 * sizeof(_Float16));
    float*    meanb = (float*)alloc(64 * sizeof(float));
    float*    invb  = (float*)alloc(64 * sizeof(float));
    float*    vals_cm  = (float*)alloc((size_t)NNZ * sizeof(float));
    float*    rs_cm    = (float*)alloc((size_t)Npix * sizeof(float));
    float*    rs_mat   = (float*)alloc((size_t)Npix * sizeof(float));
    float*    rs_cs    = (float*)alloc((size_t)Npix * sizeof(float));
    float*    loc_vals = (float*)alloc((size_t)Kloc * 9 * sizeof(float));
    float*    iu_vals  = (float*)alloc((size_t)Kiu * 5 * sizeof(float));
    float*    diag     = (float*)alloc((size_t)Npix * sizeof(float));
    float*    rv       = (float*)alloc((size_t)Npix * sizeof(float));
    float*    pv       = (float*)alloc((size_t)Npix * sizeof(float));
    float*    Apv      = (float*)alloc((size_t)Npix * sizeof(float));
    float*    zv       = (float*)alloc((size_t)Npix * sizeof(float));
    float*    Sv       = (float*)alloc((size_t)Npix * sizeof(float));
    float*    scal     = (float*)alloc(4 * sizeof(float));

    float* xv = (float*)d_out;                   // CG solution lives in d_out

    const float* CMw  = w4 + 0 * (size_t)Npix;
    const float* LOCw = w4 + 1 * (size_t)Npix;
    const float* IUw  = w4 + 2 * (size_t)Npix;
    const float* KUw  = w4 + 3 * (size_t)Npix;

    auto g = [](int n) { return (n + TB - 1) / TB; };
    const int pixTiles = Npix / 16;

    // ---------------- CNN ----------------
    k_to_f16<<<g(3 * Npix), TB, 0, stream>>>(image, act16, 3 * Npix);

    k_pack_w<<<g(4 * 1 * 512),  TB, 0, stream>>>(conv1_w, Bpk1, 3,  64, 1,  4);
    k_pack_w<<<g(4 * 18 * 512), TB, 0, stream>>>(conv2_w, Bpk2, 64, 64, 18, 4);
    k_pack_w<<<g(1 * 18 * 512), TB, 0, stream>>>(conv3_w, Bpk3, 64, 4,  18, 1);

    // layer 1: conv(3->64) + BN + lrelu
    k_conv_wmma<3, 1, 4><<<pixTiles, 128, 0, stream>>>(act16, Bpk1, conv1_b, yb, H, W, 64);
    k_bn_stats<<<64, TB, 0, stream>>>(yb, meanb, invb, Npix);
    k_bn_act<<<dim3(g(Npix), 64), TB, 0, stream>>>(yb, meanb, invb, bn1_g, bn1_b, act16, Npix);

    // layer 2: conv(64->64) + BN + lrelu
    k_conv_wmma<64, 18, 4><<<pixTiles, 128, 0, stream>>>(act16, Bpk2, conv2_b, yb, H, W, 64);
    k_bn_stats<<<64, TB, 0, stream>>>(yb, meanb, invb, Npix);
    k_bn_act<<<dim3(g(Npix), 64), TB, 0, stream>>>(yb, meanb, invb, bn2_g, bn2_b, act16, Npix);

    // layer 3: conv(64->4) -> 4 per-pixel weight maps
    k_conv_wmma<64, 18, 1><<<pixTiles, 32, 0, stream>>>(act16, Bpk3, conv3_b, w4, H, W, 4);

    // ---------------- sparse operator setup ----------------
    k_zero<<<g(Npix), TB, 0, stream>>>(rs_cm, Npix);
    k_zero<<<g(Npix), TB, 0, stream>>>(rs_mat, Npix);
    k_zero<<<g(Npix), TB, 0, stream>>>(rs_cs, Npix);
    k_zero<<<1, 32, 0, stream>>>(scal, 4);

    k_cm_setup<<<g(NNZ), TB, 0, stream>>>(Wcm_row, Wcm_data, CMw, vals_cm, rs_cm, NNZ);
    k_loc_setup<<<g(Kloc * 9), TB, 0, stream>>>(LOC_inInd, LOC_flows, LOCw, loc_vals, rs_mat, Kloc, W);
    k_iu_setup<<<g(Kiu * 5), TB, 0, stream>>>(IU_inInd, IU_flows, IU_neighInd, IUw, iu_vals, rs_cs, Kiu);

    k_cg_init<<<g(Npix), TB, 0, stream>>>(KUw, kToUconf, known, kToU, lmbda,
                                          diag, xv, rv, pv, scal, Npix);

    // ---------------- CG loop (fixed 50 iterations) ----------------
    for (int it = 0; it < CG_STEPS; ++it) {
        k_iter_begin<<<g(Npix), TB, 0, stream>>>(Sv, scal, Npix);
        // Lcm0(p): S[row] += vals * p[col]
        k_scatter_cm<<<g(NNZ), TB, 0, stream>>>(Wcm_row, Wcm_col, vals_cm, pv, Sv, NNZ);
        k_zmk<<<g(Npix), TB, 0, stream>>>(rs_cm, pv, Sv, zv, Npix);
        // Lcm0T(z): S[col] += vals * z[row]
        k_scatter_cm<<<g(NNZ), TB, 0, stream>>>(Wcm_col, Wcm_row, vals_cm, zv, Sv, NNZ);
        k_scatter_mat<<<g(Kloc * 9), TB, 0, stream>>>(LOC_inInd, loc_vals, pv, Sv, Kloc, W);
        k_scatter_cs<<<g(Kiu * 5), TB, 0, stream>>>(IU_inInd, IU_neighInd, iu_vals, pv, Sv, Kiu);
        k_combine<<<g(Npix), TB, 0, stream>>>(rs_cm, rs_mat, rs_cs, diag, zv, pv, Sv,
                                              Apv, scal, Npix);
        k_update1<<<g(Npix), TB, 0, stream>>>(Apv, pv, xv, rv, scal, Npix);
        k_update2<<<g(Npix), TB, 0, stream>>>(rv, pv, scal, Npix);
    }
}
